// AdaptedGatedAttentionWithoutqkv_6665789243991
// MI455X (gfx1250) — compile-verified, matrix operand order checked by reading, not run on hardware
//
#include <hip/hip_runtime.h>
#include <hip/hip_bf16.h>
#include <math.h>

#define NB 8
#define LDQ 1024
#define LMK 1024
#define DD 1024
#define HH 16
#define HS 64
#define KSTRIDE 72   // bf16 elements per LDS row: 144B = 9*16B (b128-aligned rows, conflict-breaking)

typedef __attribute__((ext_vector_type(16))) __bf16 v16bf;
typedef __attribute__((ext_vector_type(8)))  float  v8f;
typedef __attribute__((ext_vector_type(4)))  unsigned u32x4;
typedef __attribute__((ext_vector_type(8)))  int      i32x8;
typedef __attribute__((ext_vector_type(4)))  int      i32x4;

#if __has_builtin(__builtin_amdgcn_tensor_load_to_lds)
#define HAVE_TDM 1
#else
#define HAVE_TDM 0
#endif

union FragA {
    v16bf v;
    uint4 q[2];
    unsigned u[8];
};

static __device__ __forceinline__ unsigned short f2bf(float f) {
    unsigned u = __float_as_uint(f);
    u += 0x7FFFu + ((u >> 16) & 1u);   // round-to-nearest-even
    return (unsigned short)(u >> 16);
}
static __device__ __forceinline__ unsigned pk2(float a, float b) {
#if __has_builtin(__builtin_amdgcn_cvt_pk_bf16_f32)
    return __builtin_bit_cast(unsigned, __builtin_amdgcn_cvt_pk_bf16_f32(a, b));
#else
    return (unsigned)f2bf(a) | ((unsigned)f2bf(b) << 16);
#endif
}
static __device__ __forceinline__ v8f zero8() {
    v8f z;
#pragma unroll
    for (int i = 0; i < 8; ++i) z[i] = 0.f;
    return z;
}

// ---------------------------------------------------------------------------
// Pre-pass: bulk f32 -> bf16 conversion (8 elements / thread, fully coalesced).
// ---------------------------------------------------------------------------
__global__ __launch_bounds__(256) void cvt_bf16_kernel(const float* __restrict__ src,
                                                       unsigned short* __restrict__ dst,
                                                       int n8)
{
    const int i = blockIdx.x * 256 + threadIdx.x;
    if (i >= n8) return;
    const float4* p = (const float4*)src + (size_t)i * 2;
    float4 a = p[0], b = p[1];
    uint4 o;
    o.x = pk2(a.x, a.y); o.y = pk2(a.z, a.w);
    o.z = pk2(b.x, b.y); o.w = pk2(b.z, b.w);
    ((uint4*)dst)[i] = o;
}

// ---------------------------------------------------------------------------
// Kernel 1: flash attention. One workgroup (4 waves) per (64-query block, head,
// batch). K=V tiles of 64 keys stream through LDS: row-major tile via the
// Tensor Data Mover (bf16 byte copy, padded to 72-elem rows), transposed tile
// built cooperatively for the P*V B-operands.
// ---------------------------------------------------------------------------
__global__ __launch_bounds__(128) void attn_kernel(
    const unsigned short* __restrict__ inpBF, const unsigned short* __restrict__ memBF,
    const float* __restrict__ mask, unsigned short* __restrict__ ctx)
{
    __shared__ __align__(16) unsigned short sK [64 * KSTRIDE];    // K[key][hs]
    __shared__ __align__(16) unsigned short sKt[64 * KSTRIDE];    // Kt[hs][key]
    __shared__ __align__(16) unsigned short sP [4][16 * KSTRIDE]; // per-wave P scratch

    const int tid  = threadIdx.x;
    const int wave = tid >> 5;
    const int lane = tid & 31;
    const int half = lane >> 4;
    const int l15  = lane & 15;
    const int roff = half ? 8 : 0;

    const int qbase = blockIdx.x * 64;
    const int h     = blockIdx.y;
    const int b     = blockIdx.z;

    // ---- Q fragments (A-layout): lane = row, two contiguous 8-wide K runs ----
    const int qrow = qbase + wave * 16 + l15;
    const unsigned short* qptr = inpBF + ((size_t)b * LDQ + qrow) * DD + h * HS;
    FragA qa[2];
#pragma unroll
    for (int kc = 0; kc < 2; ++kc) {
        const unsigned short* base = qptr + kc * 32 + roff;
        qa[kc].q[0] = *(const uint4*)(base);
        qa[kc].q[1] = *(const uint4*)(base + 16);
    }

    v8f o[4];
#pragma unroll
    for (int dt = 0; dt < 4; ++dt) o[dt] = zero8();
    float mrow[8], lrow[8];
#pragma unroll
    for (int r = 0; r < 8; ++r) { mrow[r] = -3.0e38f; lrow[r] = 0.f; }

    const float scale = 0.125f;       // 1/sqrt(64)
    const int key0 = tid >> 1;        // cooperative transpose: 2 threads per key row
    const int dof  = (tid & 1) * 32;
    const unsigned ldsK = (unsigned)(uintptr_t)&sK[0];  // LDS byte offset
    (void)ldsK;

    for (int kv = 0; kv < LMK; kv += 64) {
        __syncthreads();   // previous tile fully consumed

#if HAVE_TDM
        if (wave == 0) {
            // --- Tensor Data Mover: 64x64 bf16 tile, global -> LDS (padded rows) ---
            const unsigned long long ga =
                (unsigned long long)(uintptr_t)(memBF + ((size_t)b * LMK + kv) * DD + h * HS);
            u32x4 g0;
            g0[0] = 1u;                                             // count=1, user descriptor
            g0[1] = ldsK;                                           // lds_addr
            g0[2] = (unsigned)ga;                                   // global_addr[31:0]
            g0[3] = (unsigned)((ga >> 32) & 0x01FFFFFFu) | (2u << 30); // addr[56:32] | type=2
            i32x8 g1;
            g1[0] = (int)((1u << 16)        // data_size = 2 bytes
                        | (1u << 20)        // pad_enable
                        | (4u << 22)        // pad_interval = 32 DWORDs (one 64-elem row)
                        | (3u << 25));      // pad_amount   = 4 DWORDs  (-> 72-elem stride)
            g1[1] = (int)((DD & 0xFFFFu) << 16);                    // tensor_dim0 lo16
            g1[2] = (int)((DD >> 16) & 0xFFFFu);                    // dim0 hi | dim1 lo16 (=0)
            g1[3] = (int)(1u | (64u << 16));                        // dim1 hi (65536) | tile_dim0=64
            g1[4] = (int)(64u);                                     // tile_dim1=64, tile_dim2=0
            g1[5] = (int)(DD);                                      // tensor_dim0_stride = 1024
            g1[6] = 0; g1[7] = 0;
            i32x4 z4 = {0, 0, 0, 0};
            i32x8 z8 = {0, 0, 0, 0, 0, 0, 0, 0};
            __builtin_amdgcn_tensor_load_to_lds(g0, g1, z4, z4, z8, 0);
        }
#endif
        // --- cooperative transposed tile (and row-major fallback without TDM) ---
        const unsigned short* krow =
            memBF + ((size_t)b * LMK + kv + key0) * DD + h * HS + dof;
#pragma unroll
        for (int i = 0; i < 4; ++i) {
            uint4 u = *(const uint4*)(krow + i * 8);
            const int dd = dof + i * 8;
#if !HAVE_TDM
            *(uint4*)(&sK[key0 * KSTRIDE + dd]) = u;
#endif
            sKt[(dd + 0) * KSTRIDE + key0] = (unsigned short)(u.x);
            sKt[(dd + 1) * KSTRIDE + key0] = (unsigned short)(u.x >> 16);
            sKt[(dd + 2) * KSTRIDE + key0] = (unsigned short)(u.y);
            sKt[(dd + 3) * KSTRIDE + key0] = (unsigned short)(u.y >> 16);
            sKt[(dd + 4) * KSTRIDE + key0] = (unsigned short)(u.z);
            sKt[(dd + 5) * KSTRIDE + key0] = (unsigned short)(u.z >> 16);
            sKt[(dd + 6) * KSTRIDE + key0] = (unsigned short)(u.w);
            sKt[(dd + 7) * KSTRIDE + key0] = (unsigned short)(u.w >> 16);
        }
        if (kv + 64 < LMK) __builtin_prefetch(krow + (size_t)64 * DD, 0, 1);

#if HAVE_TDM
        if (wave == 0) {
#if __has_builtin(__builtin_amdgcn_s_wait_tensorcnt)
            __builtin_amdgcn_s_wait_tensorcnt(0);
#else
            asm volatile("s_wait_tensorcnt 0x0" ::: "memory");
#endif
        }
#endif
        __syncthreads();

        // additive key-padding mask per column tile
        float madd[4];
#pragma unroll
        for (int nt = 0; nt < 4; ++nt)
            madd[nt] = -1.0e30f * (1.0f - mask[(size_t)b * LMK + kv + nt * 16 + l15]);

        // ---- S = Q * K^T : batch 4 B-fragments per K-chunk, then 4 WMMAs ----
        v8f S[4];
#pragma unroll
        for (int nt = 0; nt < 4; ++nt) S[nt] = zero8();
#pragma unroll
        for (int kc = 0; kc < 2; ++kc) {
            FragA bfr[4];
#pragma unroll
            for (int nt = 0; nt < 4; ++nt) {
                const unsigned short* base = &sK[(nt * 16 + l15) * KSTRIDE + kc * 32 + roff];
                bfr[nt].q[0] = *(const uint4*)(base);
                bfr[nt].q[1] = *(const uint4*)(base + 16);
            }
#pragma unroll
            for (int nt = 0; nt < 4; ++nt)
                S[nt] = __builtin_amdgcn_wmma_f32_16x16x32_bf16(
                    false, qa[kc].v, false, bfr[nt].v, (short)0, S[nt], false, false);
        }

        // ---- online softmax (rows live across 16-lane halves) ----
#pragma unroll
        for (int r = 0; r < 8; ++r) {
            float s0 = S[0][r] * scale + madd[0];
            float s1 = S[1][r] * scale + madd[1];
            float s2 = S[2][r] * scale + madd[2];
            float s3 = S[3][r] * scale + madd[3];
            float mx = fmaxf(fmaxf(s0, s1), fmaxf(s2, s3));
#pragma unroll
            for (int off = 1; off < 16; off <<= 1)
                mx = fmaxf(mx, __shfl_xor(mx, off, 32));
            const float mnew  = fmaxf(mrow[r], mx);
            const float alpha = __expf(mrow[r] - mnew);
            mrow[r] = mnew;
            float p0 = __expf(s0 - mnew), p1 = __expf(s1 - mnew);
            float p2 = __expf(s2 - mnew), p3 = __expf(s3 - mnew);
            float ps = p0 + p1 + p2 + p3;
#pragma unroll
            for (int off = 1; off < 16; off <<= 1)
                ps += __shfl_xor(ps, off, 32);
            lrow[r] = lrow[r] * alpha + ps;
#pragma unroll
            for (int dt = 0; dt < 4; ++dt) o[dt][r] *= alpha;
            unsigned short* prow = &sP[wave][(r + half * 8) * KSTRIDE];
            prow[ 0 + l15] = f2bf(p0);
            prow[16 + l15] = f2bf(p1);
            prow[32 + l15] = f2bf(p2);
            prow[48 + l15] = f2bf(p3);
        }
        __builtin_amdgcn_wave_barrier();
        asm volatile("s_wait_dscnt 0" ::: "memory");   // wave-private sP round trip

        // ---- re-read P in A-layout ----
        FragA pa[2];
#pragma unroll
        for (int kc = 0; kc < 2; ++kc) {
            const unsigned short* base = &sP[wave][l15 * KSTRIDE + kc * 32 + roff];
            pa[kc].q[0] = *(const uint4*)(base);
            pa[kc].q[1] = *(const uint4*)(base + 16);
        }
        // ---- O += P * V (V^T rows give contiguous key runs per d-column) ----
#pragma unroll
        for (int kc = 0; kc < 2; ++kc) {
            FragA vfr[4];
#pragma unroll
            for (int dt = 0; dt < 4; ++dt) {
                const unsigned short* base = &sKt[(dt * 16 + l15) * KSTRIDE + kc * 32 + roff];
                vfr[dt].q[0] = *(const uint4*)(base);
                vfr[dt].q[1] = *(const uint4*)(base + 16);
            }
#pragma unroll
            for (int dt = 0; dt < 4; ++dt)
                o[dt] = __builtin_amdgcn_wmma_f32_16x16x32_bf16(
                    false, pa[kc].v, false, vfr[dt].v, (short)0, o[dt], false, false);
        }
    }

    // ---- normalize + write ctx as bf16 ----
#pragma unroll
    for (int r = 0; r < 8; ++r) {
        const float inv = 1.0f / lrow[r];
        const int row = qbase + wave * 16 + r + half * 8;
        unsigned short* crow = ctx + ((size_t)b * LDQ + row) * DD + h * HS;
#pragma unroll
        for (int dt = 0; dt < 4; ++dt)
            crow[dt * 16 + l15] = f2bf(o[dt][r] * inv);
    }
}

// ---------------------------------------------------------------------------
// Kernel 2: out = sigmoid(x)*tanh(x), x = [input | ctx] @ Wc^T + bc.
// 128x64 tile per workgroup: each wave does two 16-row tiles so B-fragments
// are reused twice. All operands already bf16 -> pure b128 loads + WMMA.
// ---------------------------------------------------------------------------
__global__ __launch_bounds__(128) void gemm_gate_kernel(
    const unsigned short* __restrict__ inpBF, const unsigned short* __restrict__ ctx,
    const unsigned short* __restrict__ WcBF, const float* __restrict__ bc,
    float* __restrict__ out)
{
    const int tid  = threadIdx.x;
    const int wave = tid >> 5;
    const int lane = tid & 31;
    const int half = lane >> 4;
    const int l15  = lane & 15;
    const int roff = half ? 8 : 0;

    const int jbase = blockIdx.x * 64;
    const int mbase = blockIdx.y * 128 + wave * 16;

    const int arow = mbase + l15;      // arow = b*LDQ + q (rows contiguous over batch)
    const unsigned short* a0i = inpBF + (size_t)arow * DD;
    const unsigned short* a0c = ctx   + (size_t)arow * DD;
    const unsigned short* a1i = a0i + (size_t)64 * DD;
    const unsigned short* a1c = a0c + (size_t)64 * DD;

    v8f acc0[4], acc1[4];
#pragma unroll
    for (int nt = 0; nt < 4; ++nt) { acc0[nt] = zero8(); acc1[nt] = zero8(); }

    auto step = [&](const unsigned short* s0, const unsigned short* s1, int c) {
        FragA af0, af1;
        af0.q[0] = *(const uint4*)(s0 + roff);
        af0.q[1] = *(const uint4*)(s0 + roff + 16);
        af1.q[0] = *(const uint4*)(s1 + roff);
        af1.q[1] = *(const uint4*)(s1 + roff + 16);
        FragA bfr[4];
#pragma unroll
        for (int nt = 0; nt < 4; ++nt) {
            const unsigned short* wrow =
                WcBF + (size_t)(jbase + nt * 16 + l15) * (2 * DD) + c + roff;
            bfr[nt].q[0] = *(const uint4*)(wrow);
            bfr[nt].q[1] = *(const uint4*)(wrow + 16);
        }
#pragma unroll
        for (int nt = 0; nt < 4; ++nt) {
            acc0[nt] = __builtin_amdgcn_wmma_f32_16x16x32_bf16(
                false, af0.v, false, bfr[nt].v, (short)0, acc0[nt], false, false);
            acc1[nt] = __builtin_amdgcn_wmma_f32_16x16x32_bf16(
                false, af1.v, false, bfr[nt].v, (short)0, acc1[nt], false, false);
        }
    };

    for (int c = 0; c < DD; c += 32) step(a0i + c, a1i + c, c);            // input half
    for (int c = 0; c < DD; c += 32) step(a0c + c, a1c + c, c + DD);       // ctx half

    // ---- bias + sigmoid(x)*tanh(x) epilogue, two row tiles ----
#pragma unroll
    for (int nt = 0; nt < 4; ++nt) {
        const int j = jbase + nt * 16 + l15;
        const float bias = bc[j];
#pragma unroll
        for (int r = 0; r < 8; ++r) {
            const int row = mbase + r + half * 8;
            {
                const float x  = acc0[nt][r] + bias;
                const float sg = 1.0f / (1.0f + __expf(-x));
                out[(size_t)row * DD + j] = sg * tanhf(x);
            }
            {
                const float x  = acc1[nt][r] + bias;
                const float sg = 1.0f / (1.0f + __expf(-x));
                out[(size_t)(row + 64) * DD + j] = sg * tanhf(x);
            }
        }
    }
}

extern "C" void kernel_launch(void* const* d_in, const int* in_sizes, int n_in,
                              void* d_out, int out_size, void* d_ws, size_t ws_size,
                              hipStream_t stream) {
    const float* inp  = (const float*)d_in[0];   // [8,1024,1024]
    const float* mem  = (const float*)d_in[1];   // [8,1024,1024]
    const float* mask = (const float*)d_in[2];   // [8,1024]
    const float* Wc   = (const float*)d_in[3];   // [1024,2048]
    const float* bc   = (const float*)d_in[4];   // [1024]
    float* out = (float*)d_out;                  // [8,1024,1024] f32

    // workspace layout (bf16 elements)
    const size_t ETOK = (size_t)NB * LDQ * DD;   // 8M elements
    unsigned short* ctx   = (unsigned short*)d_ws;        // 16 MB
    unsigned short* memBF = ctx   + ETOK;                 // 16 MB
    unsigned short* inpBF = memBF + ETOK;                 // 16 MB
    unsigned short* WcBF  = inpBF + ETOK;                 // 4 MB

    // pre-pass conversions (memory-bound, ~108 MB total traffic)
    {
        const int n8i = (int)(ETOK / 8);
        cvt_bf16_kernel<<<dim3((n8i + 255) / 256), dim3(256), 0, stream>>>(inp, inpBF, n8i);
        cvt_bf16_kernel<<<dim3((n8i + 255) / 256), dim3(256), 0, stream>>>(mem, memBF, n8i);
        const int n8w = (int)((size_t)DD * 2 * DD / 8);
        cvt_bf16_kernel<<<dim3((n8w + 255) / 256), dim3(256), 0, stream>>>(Wc, WcBF, n8w);
    }

    dim3 gA(LDQ / 64, HH, NB);                   // (16,16,8) blocks of 128 threads
    attn_kernel<<<gA, dim3(128), 0, stream>>>(inpBF, memBF, mask, ctx);

    dim3 gB(DD / 64, (NB * LDQ) / 128);          // (16,64) blocks of 128 threads
    gemm_gate_kernel<<<gB, dim3(128), 0, stream>>>(inpBF, ctx, WcBF, bc, out);
}
